// DKVMN_44495861186638
// MI455X (gfx1250) — matrix operand order repaired and checked by reading, not compiled, not run on hardware
//
#include <hip/hip_runtime.h>
#include <cstdint>
#include <cstddef>

// ---------- constants ----------
#define NUM_ITEM 2000
#define DIM_KEY  256
#define DIM_VAL  128
#define BATCH    256
#define SEQLEN   512
#define NROWS    (BATCH * SEQLEN)   // 131072

// ---------- types ----------
typedef __attribute__((ext_vector_type(16))) __bf16 bf16x16;
typedef __attribute__((ext_vector_type(8)))  __bf16 bf16x8;
typedef __attribute__((ext_vector_type(8)))  float  f32x8;

// ---------- WMMA wrapper ----------
__device__ __forceinline__ f32x8 wmma_bf16(bf16x16 a, bf16x16 b, f32x8 c) {
    // (neg_a, A, neg_b, B, c_mod, C, reuse_a, reuse_b)
    return __builtin_amdgcn_wmma_f32_16x16x32_bf16(false, a, false, b, (short)0, c, false, false);
}

// A fragment: 16x32 bf16, row-major A with leading dim lda.
// lanes 0-15: row = m0+lane, elems 0..7 = K k0..k0+7, elems 8..15 = K k0+16..k0+23
// lanes16-31: row = m0+lane-16, elems 0..7 = K k0+8..k0+15, elems 8..15 = K k0+24..k0+31
__device__ __forceinline__ bf16x16 load_a_frag(const __bf16* A, int lda, int m0, int k0, int lane) {
    int row = m0 + (lane & 15);
    int kh  = (lane >> 4) * 8;
    const __bf16* p = A + (size_t)row * lda + (k0 + kh);
    bf16x8 lo = *(const bf16x8*)(p);
    bf16x8 hi = *(const bf16x8*)(p + 16);
    bf16x16 r;
#pragma unroll
    for (int i = 0; i < 8; ++i) { r[i] = lo[i]; r[i + 8] = hi[i]; }
    return r;
}

// B fragment: 32x16 bf16 taken from W stored (N,K) row-major (i.e. B^T), leading dim ldb.
// lane holds column n = n0+(lane&15); elems 0..7 = K k0+8*(lane>>4)+0..7, elems 8..15 = K k0+16+8*(lane>>4)+0..7.
__device__ __forceinline__ bf16x16 load_b_frag(const __bf16* W, int ldb, int n0, int k0, int lane) {
    int col = n0 + (lane & 15);
    int kh  = (lane >> 4) * 8;
    const __bf16* p = W + (size_t)col * ldb + (k0 + kh);
    bf16x8 lo = *(const bf16x8*)(p);
    bf16x8 hi = *(const bf16x8*)(p + 16);
    bf16x16 r;
#pragma unroll
    for (int i = 0; i < 8; ++i) { r[i] = lo[i]; r[i + 8] = hi[i]; }
    return r;
}

// ---------- kernel 0: f32 -> bf16 convert (weights) ----------
__global__ void cvt_bf16_kernel(const float* __restrict__ src, __bf16* __restrict__ dst, int n) {
    int i = blockIdx.x * blockDim.x + threadIdx.x;
    if (i < n) dst[i] = (__bf16)src[i];
}

// ---------- kernel 1: gather + convert embeddings ----------
__global__ __launch_bounds__(256)
void gather_kernel(const int* __restrict__ item, const int* __restrict__ corr,
                   const float* __restrict__ k_emb, const float* __restrict__ v_emb,
                   __bf16* __restrict__ kbf, __bf16* __restrict__ vbf) {
    int row = blockIdx.x;       // 0..NROWS-1
    int d   = threadIdx.x;      // 0..255
    int it  = item[row];
    int x   = it + NUM_ITEM * corr[row];
    kbf[(size_t)row * DIM_KEY + d] = (__bf16)k_emb[(size_t)it * DIM_KEY + d];
    vbf[(size_t)row * DIM_KEY + d] = (__bf16)v_emb[(size_t)x  * DIM_KEY + d];
}

// ---------- kernel 2: w = softmax(k @ Mk^T)  (N x 128), fused GEMM+softmax ----------
// block = 256 threads = 8 waves; each wave one 16x16 N-tile; M-strip of 16 rows.
__global__ __launch_bounds__(256)
void w_kernel(const __bf16* __restrict__ kbf, const __bf16* __restrict__ Mkbf,
              float* __restrict__ wbuf) {
    __shared__ float sl[16 * 128];
    int wave = threadIdx.x >> 5;
    int lane = threadIdx.x & 31;
    int m0 = blockIdx.x * 16;
    int n0 = wave * 16;
    f32x8 acc = {};
#pragma unroll
    for (int k0 = 0; k0 < DIM_KEY; k0 += 32) {
        bf16x16 a = load_a_frag(kbf, DIM_KEY, m0, k0, lane);
        bf16x16 b = load_b_frag(Mkbf, DIM_KEY, n0, k0, lane);
        acc = wmma_bf16(a, b, acc);
    }
    int col = n0 + (lane & 15);
    int rb  = (lane >> 4) * 8;
#pragma unroll
    for (int j = 0; j < 8; ++j) sl[(rb + j) * 128 + col] = acc[j];
    __syncthreads();

    // softmax over 128 per row: 16 threads per row, 8 cols each
    int r = threadIdx.x >> 4;
    int c = threadIdx.x & 15;
    float vals[8];
    float mx = -3.0e38f;
#pragma unroll
    for (int i = 0; i < 8; ++i) { vals[i] = sl[r * 128 + c + i * 16]; mx = fmaxf(mx, vals[i]); }
#pragma unroll
    for (int off = 8; off >= 1; off >>= 1) mx = fmaxf(mx, __shfl_xor(mx, off, 16));
    float sum = 0.f;
#pragma unroll
    for (int i = 0; i < 8; ++i) { vals[i] = __expf(vals[i] - mx); sum += vals[i]; }
#pragma unroll
    for (int off = 8; off >= 1; off >>= 1) sum += __shfl_xor(sum, off, 16);
    float inv = 1.0f / sum;
    size_t orow = (size_t)(m0 + r);
#pragma unroll
    for (int i = 0; i < 8; ++i) wbuf[orow * DIM_VAL + c + i * 16] = vals[i] * inv;
}

// ---------- kernel 3: e = sigmoid(v@eW^T+eb), a = tanh(v@aW^T+ab), both bf16 out ----------
// one wave per (mtile, ntile), computing BOTH e and a tiles (shared A fragment).
__global__ __launch_bounds__(256)
void ea_kernel(const __bf16* __restrict__ vbf,
               const __bf16* __restrict__ eWbf, const __bf16* __restrict__ aWbf,
               const float* __restrict__ e_b, const float* __restrict__ a_b,
               __bf16* __restrict__ ebf, __bf16* __restrict__ abf) {
    int gwave = (blockIdx.x * blockDim.x + threadIdx.x) >> 5;
    int lane  = threadIdx.x & 31;
    int mt = gwave >> 4;        // 16 N-tiles per 16-row strip (N=256)
    int nt = gwave & 15;
    int m0 = mt * 16, n0 = nt * 16;
    f32x8 acc_e = {}, acc_a = {};
#pragma unroll
    for (int k0 = 0; k0 < DIM_KEY; k0 += 32) {
        bf16x16 a  = load_a_frag(vbf, DIM_KEY, m0, k0, lane);
        bf16x16 be = load_b_frag(eWbf, DIM_KEY, n0, k0, lane);
        bf16x16 ba = load_b_frag(aWbf, DIM_KEY, n0, k0, lane);
        acc_e = wmma_bf16(a, be, acc_e);
        acc_a = wmma_bf16(a, ba, acc_a);
    }
    int col = n0 + (lane & 15);
    int rb  = (lane >> 4) * 8;
    float eb = e_b[col], ab = a_b[col];
#pragma unroll
    for (int j = 0; j < 8; ++j) {
        size_t idx = (size_t)(m0 + rb + j) * DIM_KEY + col;
        float ev = acc_e[j] + eb;
        float av = acc_a[j] + ab;
        ebf[idx] = (__bf16)(1.0f / (1.0f + __expf(-ev)));
        abf[idx] = (__bf16)tanhf(av);
    }
}

// ---------- kernel 4: sequential memory scan; one workgroup per batch ----------
// Register-resident state: 1024 threads = 4 vh-groups x 256 columns.
// Thread (vh, k) owns Mv[vh*32 .. vh*32+31][k] in 32 VGPRs.
// w[v] broadcast via readlane from a per-lane register (coalesced 128B load per wave).
// Per-t read-reduction of 4 vh-partials via small double-buffered LDS; 1 barrier per t.
__global__ __launch_bounds__(1024)
void scan_kernel(const float* __restrict__ wbuf,
                 const __bf16* __restrict__ ebf, const __bf16* __restrict__ abf,
                 const float* __restrict__ Mv0, __bf16* __restrict__ readsbf) {
    __shared__ float partial[2][4][DIM_KEY];   // 8 KB, double-buffered
    int tid  = threadIdx.x;
    int lane = tid & 31;
    int k    = tid & 255;
    int vh   = tid >> 8;        // 0..3
    int b    = blockIdx.x;

    float mreg[32];
#pragma unroll
    for (int v = 0; v < 32; ++v)
        mreg[v] = Mv0[(size_t)(vh * 32 + v) * DIM_KEY + k];

    for (int t = 0; t < SEQLEN; ++t) {
        size_t row = (size_t)b * SEQLEN + t;
        // this wave needs w[vh*32 .. vh*32+31]; lane l holds w[vh*32 + l]
        float wreg = wbuf[row * DIM_VAL + vh * 32 + lane];
        float ek = (float)ebf[row * DIM_KEY + k];
        float ak = (float)abf[row * DIM_KEY + k];
        float r = 0.f;
#pragma unroll
        for (int v = 0; v < 32; ++v) {
            float wv = __int_as_float(__builtin_amdgcn_readlane(__float_as_int(wreg), v));
            float m  = mreg[v];
            r = fmaf(wv, m, r);
            // m*(1 - wv*ek) + wv*ak  ==  m - m*(wv*ek) + wv*ak
            float t1 = wv * ek;
            mreg[v] = fmaf(-m, t1, fmaf(wv, ak, m));
        }
        int buf = t & 1;
        partial[buf][vh][k] = r;
        __syncthreads();
        if (tid < DIM_KEY) {
            float s = partial[buf][0][tid] + partial[buf][1][tid] +
                      partial[buf][2][tid] + partial[buf][3][tid];
            readsbf[row * DIM_KEY + tid] = (__bf16)s;
        }
        // double buffer: next t writes buf^1, so no barrier needed before next write
    }
}

// ---------- kernel 5: f = tanh([reads,k]@fW^T+fb); out = sigmoid(f.p_W + p_b) ----------
// block handles a 16-row strip; 8 waves each hold TWO N-tile accumulators (N=256),
// sharing one A fragment per K-step; p-dot fused via LDS atomics.
__global__ __launch_bounds__(256)
void f_kernel(const __bf16* __restrict__ readsbf, const __bf16* __restrict__ kbf,
              const __bf16* __restrict__ fWbf, const float* __restrict__ f_b,
              const float* __restrict__ p_W, const float* __restrict__ p_b,
              float* __restrict__ out) {
    __shared__ float rowsum[16];
    int wave = threadIdx.x >> 5;
    int lane = threadIdx.x & 31;
    int m0 = blockIdx.x * 16;
    if (threadIdx.x < 16) rowsum[threadIdx.x] = 0.f;
    __syncthreads();

    int n0a = wave * 16;
    int n0b = (wave + 8) * 16;
    f32x8 acc0 = {}, acc1 = {};
#pragma unroll
    for (int k0 = 0; k0 < 2 * DIM_KEY; k0 += 32) {
        const __bf16* A = (k0 < DIM_KEY) ? readsbf : kbf;
        int kk = (k0 < DIM_KEY) ? k0 : (k0 - DIM_KEY);
        bf16x16 a  = load_a_frag(A, DIM_KEY, m0, kk, lane);
        bf16x16 b0 = load_b_frag(fWbf, 2 * DIM_KEY, n0a, k0, lane);
        bf16x16 b1 = load_b_frag(fWbf, 2 * DIM_KEY, n0b, k0, lane);
        acc0 = wmma_bf16(a, b0, acc0);
        acc1 = wmma_bf16(a, b1, acc1);
    }
    int rb = (lane >> 4) * 8;
    {
        int col = n0a + (lane & 15);
        float fb = f_b[col], pw = p_W[col];
#pragma unroll
        for (int j = 0; j < 8; ++j) {
            float fv = tanhf(acc0[j] + fb);
            atomicAdd(&rowsum[rb + j], fv * pw);
        }
    }
    {
        int col = n0b + (lane & 15);
        float fb = f_b[col], pw = p_W[col];
#pragma unroll
        for (int j = 0; j < 8; ++j) {
            float fv = tanhf(acc1[j] + fb);
            atomicAdd(&rowsum[rb + j], fv * pw);
        }
    }
    __syncthreads();
    if (threadIdx.x < 16) {
        float s = rowsum[threadIdx.x] + p_b[0];
        out[(size_t)(m0 + threadIdx.x)] = 1.0f / (1.0f + __expf(-s));
    }
}

// ---------- host launch ----------
extern "C" void kernel_launch(void* const* d_in, const int* in_sizes, int n_in,
                              void* d_out, int out_size, void* d_ws, size_t ws_size,
                              hipStream_t stream) {
    const int*   item  = (const int*)d_in[0];
    const int*   corr  = (const int*)d_in[1];
    const float* k_emb = (const float*)d_in[2];
    const float* v_emb = (const float*)d_in[3];
    const float* Mk    = (const float*)d_in[4];
    const float* Mv0   = (const float*)d_in[5];
    const float* e_W   = (const float*)d_in[6];
    const float* e_b   = (const float*)d_in[7];
    const float* a_W   = (const float*)d_in[8];
    const float* a_b   = (const float*)d_in[9];
    const float* f_W   = (const float*)d_in[10];
    const float* f_b   = (const float*)d_in[11];
    const float* p_W   = (const float*)d_in[12];
    const float* p_b   = (const float*)d_in[13];
    float* out = (float*)d_out;

    char* ws = (char*)d_ws;
    const size_t MB64 = (size_t)64 * 1024 * 1024;
    __bf16* kbf     = (__bf16*)(ws + 0 * MB64);          // N*256 bf16 = 64MB
    __bf16* vbf     = (__bf16*)(ws + 1 * MB64);          // 64MB (reused as reads)
    __bf16* readsbf = vbf;                               // v dead after ea_kernel
    __bf16* ebf     = (__bf16*)(ws + 2 * MB64);          // 64MB
    __bf16* abf     = (__bf16*)(ws + 3 * MB64);          // 64MB
    float*  wbuf    = (float*) (ws + 4 * MB64);          // N*128 f32 = 64MB
    char*   wtail   = ws + 5 * MB64;
    __bf16* Mkbf = (__bf16*)(wtail);                     // 128*256*2 = 64KB
    __bf16* eWbf = (__bf16*)(wtail + 1 * 256 * 1024);    // 256*256*2 = 128KB
    __bf16* aWbf = (__bf16*)(wtail + 2 * 256 * 1024);
    __bf16* fWbf = (__bf16*)(wtail + 3 * 256 * 1024);    // 256*512*2 = 256KB

    // 0) convert weights to bf16
    cvt_bf16_kernel<<<(DIM_VAL * DIM_KEY + 255) / 256, 256, 0, stream>>>(Mk, Mkbf, DIM_VAL * DIM_KEY);
    cvt_bf16_kernel<<<(DIM_KEY * DIM_KEY + 255) / 256, 256, 0, stream>>>(e_W, eWbf, DIM_KEY * DIM_KEY);
    cvt_bf16_kernel<<<(DIM_KEY * DIM_KEY + 255) / 256, 256, 0, stream>>>(a_W, aWbf, DIM_KEY * DIM_KEY);
    cvt_bf16_kernel<<<(DIM_KEY * 2 * DIM_KEY + 255) / 256, 256, 0, stream>>>(f_W, fWbf, DIM_KEY * 2 * DIM_KEY);

    // 1) gather embeddings
    gather_kernel<<<NROWS, 256, 0, stream>>>(item, corr, k_emb, v_emb, kbf, vbf);

    // 2) attention weights (WMMA + fused softmax)
    w_kernel<<<NROWS / 16, 256, 0, stream>>>(kbf, Mkbf, wbuf);

    // 3) erase/add gates (WMMA)
    ea_kernel<<<(NROWS / 16) * 16 / 8, 256, 0, stream>>>(vbf, eWbf, aWbf, e_b, a_b, ebf, abf);

    // 4) sequential memory scan (register-resident state, 1024 threads/WG)
    scan_kernel<<<BATCH, 1024, 0, stream>>>(wbuf, ebf, abf, Mv0, readsbf);

    // 5) output head (WMMA + fused p-dot + sigmoid)
    f_kernel<<<NROWS / 16, 256, 0, stream>>>(readsbf, kbf, fWbf, f_b, p_W, p_b, out);
}
